// CausalGateUnit_18700287607047
// MI455X (gfx1250) — matrix operand order repaired and checked by reading, not compiled, not run on hardware
//
#include <hip/hip_runtime.h>
#include <hip/hip_bf16.h>
#include <math.h>

// Problem constants (reference: B=2, S=4096, D=512)
#define BB 2
#define SS 4096
#define DD 512
#define NROW (BB * SS)          // 8192 total rows
#define DH (DD / 2)             // 256 hidden
#define STILES (SS / 16)        // 256 s-tiles per batch

typedef __attribute__((ext_vector_type(16))) __bf16 v16bf;
typedef __attribute__((ext_vector_type(8)))  __bf16 v8bf;
typedef __attribute__((ext_vector_type(8)))  float  v8f;

// ---------------------------------------------------------------------------
// Kernel 1: per 16-row tile
//   p_pre = q@W_pre + b_pre  -> bf16   (WMMA A operand)
//   p_haz = q@W_haz + b_haz  -> bf16
//   gates = sigmoid(q@W_gate + b_gate) -> f32
//   k -> bf16                          (WMMA B operand)
// ---------------------------------------------------------------------------
__global__ __launch_bounds__(256)
void proj_kernel(const float* __restrict__ q,
                 const float* __restrict__ k,
                 const float* __restrict__ W_pre, const float* __restrict__ b_pre,
                 const float* __restrict__ W_haz, const float* __restrict__ b_haz,
                 const float* __restrict__ W_gate, const float* __restrict__ b_gate,
                 __bf16* __restrict__ p_pre_bf,
                 __bf16* __restrict__ p_haz_bf,
                 __bf16* __restrict__ k_bf,
                 float* __restrict__ gates)
{
    __shared__ float qs[16][DD];          // 32 KB
    const int tid = threadIdx.x;
    const int g0  = blockIdx.x * 16;      // first global row of this tile

    // load 16 q rows to LDS, convert 16 k rows to bf16
    for (int idx = tid; idx < 16 * DD; idx += 256) {
        const int row = idx >> 9;         // /512
        const int d   = idx & (DD - 1);
        const size_t go = (size_t)(g0 + row) * DD + d;
        qs[row][d] = q[go];
        k_bf[go]   = (__bf16)k[go];
    }
    __syncthreads();

    // each thread: 2 output columns for both projections, all 16 rows
    const int c0 = tid * 2;
    const int c1 = c0 + 1;
    float aP0[16], aP1[16], aH0[16], aH1[16];
#pragma unroll
    for (int r = 0; r < 16; ++r) { aP0[r] = 0.f; aP1[r] = 0.f; aH0[r] = 0.f; aH1[r] = 0.f; }

    for (int d = 0; d < DD; ++d) {
        const float wp0 = W_pre[d * DD + c0];
        const float wp1 = W_pre[d * DD + c1];
        const float wh0 = W_haz[d * DD + c0];
        const float wh1 = W_haz[d * DD + c1];
#pragma unroll
        for (int r = 0; r < 16; ++r) {
            const float qv = qs[r][d];
            aP0[r] = fmaf(qv, wp0, aP0[r]);
            aP1[r] = fmaf(qv, wp1, aP1[r]);
            aH0[r] = fmaf(qv, wh0, aH0[r]);
            aH1[r] = fmaf(qv, wh1, aH1[r]);
        }
    }
    const float bp0 = b_pre[c0], bp1 = b_pre[c1];
    const float bh0 = b_haz[c0], bh1 = b_haz[c1];
#pragma unroll
    for (int r = 0; r < 16; ++r) {
        const size_t o = (size_t)(g0 + r) * DD;
        p_pre_bf[o + c0] = (__bf16)(aP0[r] + bp0);
        p_pre_bf[o + c1] = (__bf16)(aP1[r] + bp1);
        p_haz_bf[o + c0] = (__bf16)(aH0[r] + bh0);
        p_haz_bf[o + c1] = (__bf16)(aH1[r] + bh1);
    }

    // gates: 16 rows x 2 gate columns = 32 dot products of length 512
    if (tid < 32) {
        const int row = tid >> 1;
        const int gi  = tid & 1;
        float acc = b_gate[gi];
        for (int d = 0; d < DD; ++d)
            acc = fmaf(qs[row][d], W_gate[d * 2 + gi], acc);
        const float g = 1.0f / (1.0f + expf(-acc));
        gates[(size_t)(g0 + row) * 2 + gi] = g;
    }
}

// ---------------------------------------------------------------------------
// Kernel 2: per (batch, 16-row s-tile) block, 8 waves.
//   waves 0-3: stream causal k-tiles vs p_pre; waves 4-7: vs p_haz.
//   16x v_wmma_f32_16x16x32_bf16 per 16x16 sim tile.
//   Interior tiles: mask-free, 8 elementwise v_max_f32 epilogue.
//   Diagonal tile: handled by one wave with per-element strict mask.
//   Scale + cross-lane row-max reduction hoisted after the t-loop.
//   Then the tiny 4->256->512 MLP in-block.
// ---------------------------------------------------------------------------
__global__ __launch_bounds__(256)
void score_mlp_kernel(const __bf16* __restrict__ p_pre_bf,
                      const __bf16* __restrict__ p_haz_bf,
                      const __bf16* __restrict__ k_bf,
                      const float* __restrict__ gates,
                      const float* __restrict__ W_s1, const float* __restrict__ b_s1,
                      const float* __restrict__ W_s2, const float* __restrict__ b_s2,
                      float* __restrict__ out)
{
    __shared__ float lds_max[2][4][16];
    __shared__ float lds_sig[2][16];
    __shared__ float lds_score[2][16];
    __shared__ float lds_h[16][DH];    // 16 KB

    const int tid   = threadIdx.x;
    const int lane  = tid & 31;
    const int wave  = tid >> 5;
    const int wsub  = wave & 3;          // t-tile strider within the 4 waves
    const int mat   = wave >> 2;         // 0 = pre, 1 = haz
    const int si    = blockIdx.x;        // s-tile index [0,256)
    const int bb    = blockIdx.y;        // batch
    const int s0    = si * 16;
    const int gbase = bb * SS + s0;      // first global row of s-tile
    const float scale = 0.044194173824159216f;   // 1/sqrt(512)

    const __bf16* pbf = mat ? p_haz_bf : p_pre_bf;

    // --- load this wave's 16x512 bf16 A-tile into registers (contiguous k order, same as B) ---
    // lane holds row M = lane&15; per-lane v16bf = K[base..base+15],
    // base += (lane>=16 ? 16 : 0)
    const int half = lane >> 4;
    const __bf16* prow = pbf + (size_t)(gbase + (lane & 15)) * DD + (half ? 16 : 0);
    v16bf a[16];
#pragma unroll
    for (int kc = 0; kc < 16; ++kc) {
        a[kc] = *(const v16bf*)(prow + kc * 32);
    }

    // --- elementwise running max (per lane, per C-VGPR slot) ---
    float run[8];
#pragma unroll
    for (int r = 0; r < 8; ++r) run[r] = -INFINITY;

    // B-operand base: lane holds k-row (column of B) t0 + (lane&15),
    // K-halves: lanes>=16 take K offset +16 within each 32-chunk.
    const __bf16* kbase_ptr = k_bf + (size_t)bb * SS * DD + (half ? 16 : 0);
    const int colsel = lane & 15;

    // --- interior tiles (ti < si): fully inside strict lower triangle ---
    for (int ti = wsub; ti < si; ti += 4) {
        const __bf16* krow = kbase_ptr + (size_t)(ti * 16 + colsel) * DD;
        v8f acc = {0.f, 0.f, 0.f, 0.f, 0.f, 0.f, 0.f, 0.f};
#pragma unroll
        for (int kc = 0; kc < 16; ++kc) {
            const v16bf bv = *(const v16bf*)(krow + kc * 32);
            acc = __builtin_amdgcn_wmma_f32_16x16x32_bf16(
                      false, a[kc], false, bv, (short)0, acc, false, false);
        }
#pragma unroll
        for (int r = 0; r < 8; ++r)
            run[r] = fmaxf(run[r], acc[r]);
    }

    // --- diagonal tile (ti == si): one wave per matrix, strict N < M mask ---
    if (wsub == (si & 3)) {                     // wave-uniform branch
        const __bf16* krow = kbase_ptr + (size_t)(si * 16 + colsel) * DD;
        v8f acc = {0.f, 0.f, 0.f, 0.f, 0.f, 0.f, 0.f, 0.f};
#pragma unroll
        for (int kc = 0; kc < 16; ++kc) {
            const v16bf bv = *(const v16bf*)(krow + kc * 32);
            acc = __builtin_amdgcn_wmma_f32_16x16x32_bf16(
                      false, a[kc], false, bv, (short)0, acc, false, false);
        }
        const int mrow = half * 8;              // row base for this lane half
#pragma unroll
        for (int r = 0; r < 8; ++r) {
            const float v = (colsel < mrow + r) ? acc[r] : -INFINITY;
            run[r] = fmaxf(run[r], v);
        }
    }

    // --- hoisted: cross-lane row-max reduction + scale (scale > 0) ---
#pragma unroll
    for (int r = 0; r < 8; ++r) {
        float v = run[r];
#pragma unroll
        for (int off = 1; off < 16; off <<= 1)  // xor 1,2,4,8 stays in half
            v = fmaxf(v, __shfl_xor(v, off, 32));
        run[r] = v * scale;
    }

    // --- cross-wave max reduce via LDS ---
    if ((lane & 15) == 0) {
#pragma unroll
        for (int r = 0; r < 8; ++r)
            lds_max[mat][wsub][half * 8 + r] = run[r];
    }
    __syncthreads();

    if (tid < 32) {
        const int m   = tid >> 4;
        const int row = tid & 15;
        float v = lds_max[m][0][row];
        v = fmaxf(v, lds_max[m][1][row]);
        v = fmaxf(v, lds_max[m][2][row]);
        v = fmaxf(v, lds_max[m][3][row]);
        if (isinf(v)) v = 0.0f;                 // where(isinf, 0)
        lds_score[m][row] = v;
        lds_sig[m][row]   = gates[(size_t)(gbase + row) * 2 + m];
    }
    __syncthreads();

    // --- h = relu(raw @ W_s1 + b_s1), raw = [g_pre, score_pre, g_haz, score_haz] ---
    for (int idx = tid; idx < 16 * DH; idx += 256) {
        const int row = idx >> 8;          // /256
        const int j   = idx & (DH - 1);
        float h = b_s1[j];
        h = fmaf(lds_sig[0][row],   W_s1[0 * DH + j], h);
        h = fmaf(lds_score[0][row], W_s1[1 * DH + j], h);
        h = fmaf(lds_sig[1][row],   W_s1[2 * DH + j], h);
        h = fmaf(lds_score[1][row], W_s1[3 * DH + j], h);
        lds_h[row][j] = fmaxf(h, 0.0f);
    }
    __syncthreads();

    // --- out = h @ W_s2 + b_s2 ---
    for (int idx = tid; idx < 16 * DD; idx += 256) {
        const int row = idx >> 9;          // /512
        const int c   = idx & (DD - 1);
        float acc = b_s2[c];
        for (int j = 0; j < DH; ++j)
            acc = fmaf(lds_h[row][j], W_s2[j * DD + c], acc);
        out[(size_t)(gbase + row) * DD + c] = acc;
    }
}

extern "C" void kernel_launch(void* const* d_in, const int* in_sizes, int n_in,
                              void* d_out, int out_size, void* d_ws, size_t ws_size,
                              hipStream_t stream) {
    // inputs (setup_inputs order): x, q, k, W_pre, b_pre, W_haz, b_haz,
    //                              W_gate, b_gate, W_s1, b_s1, W_s2, b_s2
    const float* q      = (const float*)d_in[1];
    const float* k      = (const float*)d_in[2];
    const float* W_pre  = (const float*)d_in[3];
    const float* b_pre  = (const float*)d_in[4];
    const float* W_haz  = (const float*)d_in[5];
    const float* b_haz  = (const float*)d_in[6];
    const float* W_gate = (const float*)d_in[7];
    const float* b_gate = (const float*)d_in[8];
    const float* W_s1   = (const float*)d_in[9];
    const float* b_s1   = (const float*)d_in[10];
    const float* W_s2   = (const float*)d_in[11];
    const float* b_s2   = (const float*)d_in[12];
    float* out = (float*)d_out;

    // workspace carve-up (~25.2 MB)
    __bf16* p_pre_bf = (__bf16*)d_ws;
    __bf16* p_haz_bf = p_pre_bf + (size_t)NROW * DD;
    __bf16* k_bf     = p_haz_bf + (size_t)NROW * DD;
    float*  gates    = (float*)(k_bf + (size_t)NROW * DD);

    proj_kernel<<<NROW / 16, 256, 0, stream>>>(
        q, k, W_pre, b_pre, W_haz, b_haz, W_gate, b_gate,
        p_pre_bf, p_haz_bf, k_bf, gates);

    score_mlp_kernel<<<dim3(STILES, BB), 256, 0, stream>>>(
        p_pre_bf, p_haz_bf, k_bf, gates,
        W_s1, b_s1, W_s2, b_s2, out);
}
